// TICED_84799834292987
// MI455X (gfx1250) — compile-verified, matrix operand order checked by reading, not run on hardware
//
#include <hip/hip_runtime.h>
#include <hip/hip_bf16.h>
#include <math.h>

typedef __bf16 bf16_t;
typedef __attribute__((ext_vector_type(16))) __bf16 v16bf;
typedef __attribute__((ext_vector_type(8)))  float  v8f;

#define DD 128
#define LL 50
#define BB 512
#define BLR (BB*LL)          // 25600 rows
#define NHEADS 4
#define HDIM 32
#define KTOP 12
#define LAMBDA_C 0.5f
#define BN_EPS_C 1e-5f

// ---------------- WMMA fragment helpers (ISA 7.12.2 layouts) ----------------
__device__ __forceinline__ int frag_kofs(int i, int half) {
  // element i of the v16bf fragment holds K = i + 8*(half + (i>>3))
  return i + 8 * (half + (i >> 3));
}

// Load A/B fragment from a bf16 matrix whose K axis has stride 1 and whose
// "row" (M for A, N for B when B is stored as N x K) has stride rowStride.
__device__ __forceinline__ v16bf ldfrag_bf16(const bf16_t* p, long rowStride,
                                             int row0, int rdiv, int kbase, int lane) {
  int half = lane >> 4;
  long r = (long)((row0 + (lane & 15)) / rdiv);
  const bf16_t* q = p + r * rowStride + kbase;
  v16bf f;
#pragma unroll
  for (int i = 0; i < 16; ++i) f[i] = q[frag_kofs(i, half)];
  return f;
}

__device__ __forceinline__ v16bf ldfrag_f32(const float* p, long rowStride,
                                            int row0, int rdiv, int kbase, int lane) {
  int half = lane >> 4;
  long r = (long)((row0 + (lane & 15)) / rdiv);
  const float* q = p + r * rowStride + kbase;
  v16bf f;
#pragma unroll
  for (int i = 0; i < 16; ++i) f[i] = (bf16_t)q[frag_kofs(i, half)];
  return f;
}

__device__ __forceinline__ v8f wmma_bf16(v16bf a, v16bf b, v8f c) {
  return __builtin_amdgcn_wmma_f32_16x16x32_bf16(false, a, false, b, (short)0, c, false, false);
}

__device__ __forceinline__ float sigmoidf_(float x) { return 1.0f / (1.0f + __expf(-x)); }

// ---------------- prep: per-column frequency index + mask ----------------
__global__ void k_prep_idx(const int* __restrict__ seq, int* __restrict__ itempos,
                           float* __restrict__ mask) {
  int b = blockIdx.x;
  int i = threadIdx.x;
  if (i >= LL) return;
  int v = seq[i * BB + b];
  int c = 0;
  for (int j = 0; j < LL; ++j) c += (seq[j * BB + b] == v) ? 1 : 0;
  int m = (v > 0) ? 1 : 0;
  itempos[b * LL + i] = c * m;
  mask[b * LL + i] = (float)m;
}

// ---------------- gather item + positional embeddings ----------------
__global__ void k_gather(const int* __restrict__ seq, const int* __restrict__ itempos,
                         const float* __restrict__ emb, const float* __restrict__ pos_emb,
                         float* __restrict__ embsf, bf16_t* __restrict__ embsbf,
                         bf16_t* __restrict__ posbf) {
  long idx = (long)blockIdx.x * blockDim.x + threadIdx.x;
  if (idx >= (long)BLR * DD) return;
  int d = (int)(idx & (DD - 1));
  long r = idx >> 7;                       // r = b*LL + l
  int b = (int)(r / LL), l = (int)(r % LL);
  int it = seq[l * BB + b];
  float e = emb[(long)it * DD + d];
  embsf[idx] = e;
  embsbf[idx] = (bf16_t)e;
  int ip = itempos[r];
  posbf[idx] = (bf16_t)pos_emb[ip * DD + d];
}

// ---------------- generic dual-input fused GEMM ----------------
// Y[M x N] = act( X1[M/rdiv1,128] @ W1^T + b1 (+ X2[M/rdiv2,128] @ W2^T + b2) )
// X bf16 row-major (ld=128), W f32 (N x 128) row-major. act: 0 none, 1 sigmoid, 2 tanh.
// Each wave owns a 16x64 output strip: A fragment reused across 4 wmma.
__global__ void k_gemm(const bf16_t* __restrict__ X1, int rdiv1,
                       const float* __restrict__ W1, const float* __restrict__ b1,
                       const bf16_t* __restrict__ X2, int rdiv2,
                       const float* __restrict__ W2, const float* __restrict__ b2,
                       float* __restrict__ Yf, bf16_t* __restrict__ Ybf,
                       int M, int N, int act) {
  int lane = threadIdx.x & 31;
  int wave = threadIdx.x >> 5;
  int stripsN = N >> 6;                       // 64-column strips
  int strips = (M >> 4) * stripsN;
  int s = blockIdx.x * 8 + wave;
  if (s >= strips) return;
  int tm = s / stripsN, tn = s % stripsN;
  __builtin_prefetch(W1 + (long)tn * 64 * 128, 0, 1);
  if (W2) __builtin_prefetch(W2 + (long)tn * 64 * 128, 0, 1);
  v8f acc[4] = {};
#pragma unroll
  for (int kc = 0; kc < 4; ++kc) {
    v16bf a = ldfrag_bf16(X1, 128, tm * 16, rdiv1, kc * 32, lane);
#pragma unroll
    for (int j = 0; j < 4; ++j) {
      v16bf b = ldfrag_f32(W1, 128, tn * 64 + j * 16, 1, kc * 32, lane);
      acc[j] = wmma_bf16(a, b, acc[j]);
    }
  }
  if (X2) {
#pragma unroll
    for (int kc = 0; kc < 4; ++kc) {
      v16bf a = ldfrag_bf16(X2, 128, tm * 16, rdiv2, kc * 32, lane);
#pragma unroll
      for (int j = 0; j < 4; ++j) {
        v16bf b = ldfrag_f32(W2, 128, tn * 64 + j * 16, 1, kc * 32, lane);
        acc[j] = wmma_bf16(a, b, acc[j]);
      }
    }
  }
  int half = lane >> 4;
#pragma unroll
  for (int j = 0; j < 4; ++j) {
    int col = tn * 64 + j * 16 + (lane & 15);
    float bias = (b1 ? b1[col] : 0.0f) + (b2 ? b2[col] : 0.0f);
#pragma unroll
    for (int r = 0; r < 8; ++r) {
      int row = tm * 16 + r + 8 * half;
      float v = acc[j][r] + bias;
      if (act == 1) v = sigmoidf_(v);
      else if (act == 2) v = tanhf(v);
      long o = (long)row * N + col;
      if (Yf) Yf[o] = v;
      if (Ybf) Ybf[o] = (bf16_t)v;
    }
  }
}

// ---------------- GRU: one workgroup owns 64 batch rows; Whh + h in LDS ----------------
// gx = X @ Wih^T + bih precomputed in GX (BLR x 384). Gate order [r,z,n] (torch).
__global__ void k_gru(const float* __restrict__ GX, const float* __restrict__ Whh,
                      const float* __restrict__ bhh,
                      bf16_t* __restrict__ nowbf, bf16_t* __restrict__ globf) {
  extern __shared__ char smem[];
  bf16_t* sWhh = (bf16_t*)smem;                              // 384*128 bf16
  bf16_t* sH   = (bf16_t*)(smem + 384 * 128 * 2);            // 64*128 bf16
  float*  sG   = (float*)(smem + 384 * 128 * 2 + 64 * 128 * 2); // 64*384 f32
  int tid = threadIdx.x, lane = tid & 31, wave = tid >> 5;
  int bbase = blockIdx.x * 64;
  for (int i = tid; i < 384 * 128; i += 256) sWhh[i] = (bf16_t)Whh[i];
  for (int i = tid; i < 64 * 128; i += 256) sH[i] = (bf16_t)0.0f;
  int b_loc = tid >> 2;
  int qd = (tid & 3) * 32;
  float h[32], glo[32];
#pragma unroll
  for (int j = 0; j < 32; ++j) { h[j] = 0.0f; glo[j] = 0.0f; }
  __syncthreads();
  for (int t = 0; t < LL; ++t) {
    // gh = h @ Whh^T : 64x384 = 4 x 6 strips of 16x64, 8 waves
    for (int s = wave; s < 24; s += 8) {
      int tm = s / 6, tn = s % 6;
      v8f acc[4] = {};
#pragma unroll
      for (int kc = 0; kc < 4; ++kc) {
        v16bf a = ldfrag_bf16(sH, 128, tm * 16, 1, kc * 32, lane);
#pragma unroll
        for (int j = 0; j < 4; ++j) {
          v16bf b = ldfrag_bf16(sWhh, 128, tn * 64 + j * 16, 1, kc * 32, lane);
          acc[j] = wmma_bf16(a, b, acc[j]);
        }
      }
      int half = lane >> 4;
#pragma unroll
      for (int j = 0; j < 4; ++j) {
        int col = tn * 64 + j * 16 + (lane & 15);
#pragma unroll
        for (int r = 0; r < 8; ++r) sG[(tm * 16 + r + 8 * half) * 384 + col] = acc[j][r];
      }
    }
    __syncthreads();
    long gr = ((long)(bbase + b_loc) * LL + t) * 384;
#pragma unroll
    for (int j = 0; j < 32; ++j) {
      int d = qd + j;
      float xr = GX[gr + d], xz = GX[gr + 128 + d], xn = GX[gr + 256 + d];
      float hr = sG[b_loc * 384 + d] + bhh[d];
      float hz = sG[b_loc * 384 + 128 + d] + bhh[128 + d];
      float hn = sG[b_loc * 384 + 256 + d] + bhh[256 + d];
      float rg = sigmoidf_(xr + hr);
      float zg = sigmoidf_(xz + hz);
      float ng = tanhf(xn + rg * hn);
      float hv = (1.0f - zg) * ng + zg * h[j];
      h[j] = hv;
      glo[j] += hv;
      sH[b_loc * 128 + d] = (bf16_t)hv;
    }
    __syncthreads();
  }
#pragma unroll
  for (int j = 0; j < 32; ++j) {
    long o = (long)(bbase + b_loc) * 128 + qd + j;
    nowbf[o] = (bf16_t)h[j];
    globf[o] = (bf16_t)glo[j];
  }
}

// ---------------- small elementwise kernels ----------------
__global__ void k_ew_tmp1(const float* __restrict__ user_p, const float* __restrict__ embsf,
                          bf16_t* __restrict__ outbf) {
  long idx = (long)blockIdx.x * blockDim.x + threadIdx.x;
  if (idx >= (long)BLR * DD) return;
  int d = (int)(idx & 127); long r = idx >> 7; int b = (int)(r / LL);
  outbf[idx] = (bf16_t)(user_p[b * 128 + d] * embsf[idx]);
}

__global__ void k_ew_uw(const float* __restrict__ user_p, const float* __restrict__ user_e,
                        const float* __restrict__ user_r, const float* __restrict__ uwh,
                        bf16_t* __restrict__ uwbf) {
  long idx = (long)blockIdx.x * blockDim.x + threadIdx.x;
  if (idx >= (long)BLR * DD) return;
  int d = (int)(idx & 127); long r = idx >> 7; int b = (int)(r / LL);
  float v = user_p[b * 128 + d] * user_e[b * 128 + d] + user_r[idx] * uwh[idx];
  uwbf[idx] = (bf16_t)v;
}

__global__ void k_ew_final(const float* __restrict__ fre_h, const float* __restrict__ user_e,
                           const float* __restrict__ fre_e, const float* __restrict__ pwh,
                           float* __restrict__ ef) {
  long idx = (long)blockIdx.x * blockDim.x + threadIdx.x;
  if (idx >= (long)BLR * DD) return;
  int d = (int)(idx & 127); long r = idx >> 7; int b = (int)(r / LL);
  ef[idx] = fre_h[idx] * user_e[b * 128 + d] + fre_e[idx] * pwh[idx];
}

// ---------------- BatchNorm1d over (B, D) per position l ----------------
__global__ void k_bn_stats(const float* __restrict__ ef, float* __restrict__ mu,
                           float* __restrict__ var) {
  int l = blockIdx.x;
  __shared__ float rs[256], rs2[256];
  float s = 0.0f, s2 = 0.0f;
  for (int i = threadIdx.x; i < BB * DD; i += blockDim.x) {
    int b = i >> 7, d = i & 127;
    float v = ef[((long)b * LL + l) * DD + d];
    s += v; s2 += v * v;
  }
  rs[threadIdx.x] = s; rs2[threadIdx.x] = s2; __syncthreads();
  for (int o = 128; o > 0; o >>= 1) {
    if (threadIdx.x < (unsigned)o) { rs[threadIdx.x] += rs[threadIdx.x + o]; rs2[threadIdx.x] += rs2[threadIdx.x + o]; }
    __syncthreads();
  }
  if (threadIdx.x == 0) {
    float m = rs[0] / (float)(BB * DD);
    mu[l] = m;
    var[l] = rs2[0] / (float)(BB * DD) - m * m;
  }
}

__global__ void k_bn_apply(const float* __restrict__ ef, const float* __restrict__ mu,
                           const float* __restrict__ var, float* __restrict__ F,
                           bf16_t* __restrict__ Fbf) {
  long idx = (long)blockIdx.x * blockDim.x + threadIdx.x;
  if (idx >= (long)BLR * DD) return;
  int l = (int)((idx >> 7) % LL);
  float v = (ef[idx] - mu[l]) * rsqrtf(var[l] + BN_EPS_C);
  F[idx] = v;
  Fbf[idx] = (bf16_t)v;
}

// ---------------- row dot: out[r] = sum_d X[r,d]*w[d] + b ----------------
__global__ void k_rowdot(const float* __restrict__ X, const float* __restrict__ w,
                         const float* __restrict__ bptr, float* __restrict__ out, int M) {
  int r = blockIdx.x * 8 + (threadIdx.x >> 5);
  if (r >= M) return;
  int lane = threadIdx.x & 31;
  float s = 0.0f;
  for (int d = lane; d < 128; d += 32) s += X[(long)r * 128 + d] * w[d];
#pragma unroll
  for (int o = 16; o > 0; o >>= 1) s += __shfl_xor(s, o, 32);
  if (lane == 0) out[r] = s + bptr[0];
}

// ---------------- l1[b,:] = sum_l mask*s*ini_emb ----------------
__global__ void k_l1(const float* __restrict__ mask, const float* __restrict__ svec,
                     const float* __restrict__ ini_emb, float* __restrict__ l1f,
                     bf16_t* __restrict__ l1bf) {
  int idx = blockIdx.x * blockDim.x + threadIdx.x;
  if (idx >= BB * 128) return;
  int b = idx >> 7, d = idx & 127;
  float acc = 0.0f;
  for (int l = 0; l < LL; ++l) {
    long r = (long)b * LL + l;
    acc += mask[r] * svec[r] * ini_emb[r * 128 + d];
  }
  l1f[idx] = acc;
  l1bf[idx] = (bf16_t)acc;
}

// ---------------- value projection broadcast: vp = mask*(vec@Wv^T) + bv ----------------
__global__ void k_vp(const float* __restrict__ mask, const float* __restrict__ VL,
                     const float* __restrict__ bv, bf16_t* __restrict__ vpbf) {
  long idx = (long)blockIdx.x * blockDim.x + threadIdx.x;
  if (idx >= (long)BLR * DD) return;
  int d = (int)(idx & 127); long r = idx >> 7; int b = (int)(r / LL);
  vpbf[idx] = (bf16_t)(mask[r] * VL[b * 128 + d] + bv[d]);
}

// ---------------- attention block: one WG per (l, h, 64-row tile of T=512) ----------------
__global__ void k_attn(const bf16_t* __restrict__ qp, const bf16_t* __restrict__ kp,
                       const bf16_t* __restrict__ vp, bf16_t* __restrict__ ctx,
                       float* __restrict__ colsum) {
  extern __shared__ char smem[];
  float* S = (float*)smem;                          // 64 x 512 f32
  bf16_t* P = (bf16_t*)(smem + 64 * 512 * 4);       // 64 x 512 bf16
  int blk = blockIdx.x;
  int tb = blk & 7;
  int h = (blk >> 3) & 3;
  int l = blk >> 5;
  int t0 = tb * 64;
  int lane = threadIdx.x & 31, wave = threadIdx.x >> 5;
  const float scale = 0.17677669529663687f;         // 1/sqrt(32)
  long rowStride = (long)LL * 128;
  const bf16_t* qbase = qp + (long)l * 128 + h * HDIM;
  const bf16_t* kbase = kp + (long)l * 128 + h * HDIM;
  const bf16_t* vbase = vp + (long)l * 128 + h * HDIM;
  // S = scale * Q K^T  (64 x 512) : 4 x 8 strips of 16x64, K=32 (one wmma per tile)
  for (int s = wave; s < 32; s += 8) {
    int tm = s >> 3, tn = s & 7;
    v16bf a = ldfrag_bf16(qbase, rowStride, t0 + tm * 16, 1, 0, lane);
    v8f acc[4] = {};
#pragma unroll
    for (int j = 0; j < 4; ++j) {
      v16bf b = ldfrag_bf16(kbase, rowStride, tn * 64 + j * 16, 1, 0, lane);
      acc[j] = wmma_bf16(a, b, acc[j]);
    }
    int half = lane >> 4;
#pragma unroll
    for (int j = 0; j < 4; ++j) {
      int col = tn * 64 + j * 16 + (lane & 15);
#pragma unroll
      for (int r = 0; r < 8; ++r) S[(tm * 16 + r + 8 * half) * 512 + col] = acc[j][r] * scale;
    }
  }
  __syncthreads();
  // rowwise softmax, write P (bf16)
  for (int row = wave; row < 64; row += 8) {
    float mx = -1e30f;
    for (int c = lane; c < 512; c += 32) mx = fmaxf(mx, S[row * 512 + c]);
#pragma unroll
    for (int o = 16; o > 0; o >>= 1) mx = fmaxf(mx, __shfl_xor(mx, o, 32));
    float sum = 0.0f;
    for (int c = lane; c < 512; c += 32) {
      float e = __expf(S[row * 512 + c] - mx);
      S[row * 512 + c] = e;
      sum += e;
    }
#pragma unroll
    for (int o = 16; o > 0; o >>= 1) sum += __shfl_xor(sum, o, 32);
    float inv = 1.0f / sum;
    for (int c = lane; c < 512; c += 32) P[row * 512 + c] = (bf16_t)(S[row * 512 + c] * inv);
  }
  __syncthreads();
  // partial column sums of P (for aw.sum over t), accumulated across heads/tiles
  for (int c = threadIdx.x; c < 512; c += 256) {
    float s = 0.0f;
    for (int r = 0; r < 64; ++r) s += (float)P[r * 512 + c];
    atomicAdd(&colsum[l * BB + c], s);
  }
  // ctx = P @ V' (64x512 @ 512x32): 4 x 2 tiles, one per wave, K=512
  {
    int tm = wave >> 1, tn = wave & 1;
    v8f acc = {};
    for (int kc = 0; kc < 16; ++kc) {
      v16bf a = ldfrag_bf16(P, 512, tm * 16, 1, kc * 32, lane);
      v16bf b;
      int half = lane >> 4;
#pragma unroll
      for (int i = 0; i < 16; ++i) {
        int k = kc * 32 + frag_kofs(i, half);
        b[i] = vbase[(long)k * rowStride + tn * 16 + (lane & 15)];
      }
      acc = wmma_bf16(a, b, acc);
    }
    int col = tn * 16 + (lane & 15), half = lane >> 4;
#pragma unroll
    for (int r = 0; r < 8; ++r) {
      int t = t0 + tm * 16 + r + 8 * half;
      ctx[((long)t * LL + l) * 128 + h * HDIM + col] = (bf16_t)acc[r];
    }
  }
}

// ---------------- beta pre: row dot of sigmoid(out + cs*F) with v_h2std ----------------
__global__ void k_beta_pre(const float* __restrict__ Aout, const float* __restrict__ F,
                           const float* __restrict__ colsum, const float* __restrict__ w,
                           const float* __restrict__ bptr, float* __restrict__ braw) {
  int r = blockIdx.x * 8 + (threadIdx.x >> 5);
  if (r >= BLR) return;
  int lane = threadIdx.x & 31;
  int b = r / LL, l = r % LL;
  float cs = colsum[l * BB + b] * (1.0f / NHEADS);
  float s = 0.0f;
  for (int d = lane; d < 128; d += 32) {
    float v = Aout[(long)r * 128 + d] + cs * F[(long)r * 128 + d];
    s += sigmoidf_(v) * w[d];
  }
#pragma unroll
  for (int o = 16; o > 0; o >>= 1) s += __shfl_xor(s, o, 32);
  if (lane == 0) braw[r] = s + bptr[0];
}

__global__ void k_beta_softmax(const float* __restrict__ braw, float* __restrict__ beta,
                               float* __restrict__ beta_v) {
  int b = blockIdx.x;
  int t = threadIdx.x;  // 64 threads
  __shared__ float sv[64];
  float v = (t < LL) ? braw[b * LL + t] : -1e30f;
  sv[t] = v; __syncthreads();
  for (int o = 32; o > 0; o >>= 1) { if (t < o) sv[t] = fmaxf(sv[t], sv[t + o]); __syncthreads(); }
  float mx = sv[0]; __syncthreads();
  float e = (t < LL) ? __expf(v - mx) : 0.0f;
  sv[t] = e; __syncthreads();
  for (int o = 32; o > 0; o >>= 1) { if (t < o) sv[t] += sv[t + o]; __syncthreads(); }
  float bb = e / sv[0];
  if (t < LL) beta[b * LL + t] = bb;
  if (b == 0) {
    __syncthreads();
    sv[t] = (t < LL) ? bb : 0.0f; __syncthreads();
    for (int o = 32; o > 0; o >>= 1) { if (t < o) sv[t] += sv[t + o]; __syncthreads(); }
    if (t == 0) beta_v[0] = sv[0] / (float)LL;
  }
}

__global__ void k_sess(const float* __restrict__ beta, const float* __restrict__ beta_v,
                       const float* __restrict__ F, float* __restrict__ sess,
                       bf16_t* __restrict__ sessbf) {
  int idx = blockIdx.x * blockDim.x + threadIdx.x;
  if (idx >= BB * 128) return;
  int b = idx >> 7, d = idx & 127;
  float thr = LAMBDA_C * beta_v[0];
  float s = 0.0f;
  for (int l = 0; l < LL; ++l) {
    float w = beta[b * LL + l];
    w = (w - thr > 0.0f) ? w : 0.0f;
    s += w * F[((long)b * LL + l) * 128 + d];
  }
  sess[idx] = s;
  sessbf[idx] = (bf16_t)s;
}

// ---------------- L2-normalize rows ----------------
__global__ void k_norm(const float* __restrict__ sess, bf16_t* __restrict__ out) {
  int r = blockIdx.x * 8 + (threadIdx.x >> 5);
  if (r >= BB) return;
  int lane = threadIdx.x & 31;
  float s = 0.0f;
  for (int d = lane; d < 128; d += 32) { float v = sess[r * 128 + d]; s += v * v; }
#pragma unroll
  for (int o = 16; o > 0; o >>= 1) s += __shfl_xor(s, o, 32);
  float inv = 1.0f / fmaxf(sqrtf(s), 1e-12f);
  for (int d = lane; d < 128; d += 32) out[r * 128 + d] = (bf16_t)(sess[r * 128 + d] * inv);
}

// ---------------- cosine similarity (both operands bf16, wmma, 16x64 strips) ----------------
__global__ void k_sim(const bf16_t* __restrict__ X, float* __restrict__ sim) {
  int lane = threadIdx.x & 31, wave = threadIdx.x >> 5;
  int s = blockIdx.x * 8 + wave;
  if (s >= 32 * 8) return;                  // (512/16) x (512/64) strips
  int tm = s >> 3, tn = s & 7;
  v8f acc[4] = {};
#pragma unroll
  for (int kc = 0; kc < 4; ++kc) {
    v16bf a = ldfrag_bf16(X, 128, tm * 16, 1, kc * 32, lane);
#pragma unroll
    for (int j = 0; j < 4; ++j) {
      v16bf b = ldfrag_bf16(X, 128, tn * 64 + j * 16, 1, kc * 32, lane);
      acc[j] = wmma_bf16(a, b, acc[j]);
    }
  }
  int half = lane >> 4;
#pragma unroll
  for (int j = 0; j < 4; ++j) {
    int col = tn * 64 + j * 16 + (lane & 15);
#pragma unroll
    for (int r = 0; r < 8; ++r) sim[(tm * 16 + r + 8 * half) * 512 + col] = acc[j][r];
  }
}

// ---------------- top-K + softmax mix ----------------
__global__ void k_topk(const float* __restrict__ sim, const float* __restrict__ sess,
                       float* __restrict__ outp) {
  int b = blockIdx.x;
  int t = threadIdx.x;  // 256
  __shared__ float row[512];
  __shared__ float sv[256];
  __shared__ int si[256];
  __shared__ float topv[KTOP];
  __shared__ int topi[KTOP];
  __shared__ float wk[KTOP];
  for (int c = t; c < 512; c += 256) row[c] = sim[b * 512 + c];
  __syncthreads();
  for (int k = 0; k < KTOP; ++k) {
    float bv = -1e30f; int bi = 0;
    for (int c = t; c < 512; c += 256) { if (row[c] > bv) { bv = row[c]; bi = c; } }
    sv[t] = bv; si[t] = bi; __syncthreads();
    for (int o = 128; o > 0; o >>= 1) {
      if (t < o) { if (sv[t + o] > sv[t]) { sv[t] = sv[t + o]; si[t] = si[t + o]; } }
      __syncthreads();
    }
    if (t == 0) { topv[k] = sv[0]; topi[k] = si[0]; row[si[0]] = -1e30f; }
    __syncthreads();
  }
  if (t == 0) {
    float mx = -1e30f;
    for (int k = 0; k < KTOP; ++k) mx = fmaxf(mx, topv[k]);
    float s = 0.0f;
    for (int k = 0; k < KTOP; ++k) { wk[k] = __expf(topv[k] - mx); s += wk[k]; }
    for (int k = 0; k < KTOP; ++k) wk[k] /= s;
  }
  __syncthreads();
  for (int d = t; d < 128; d += 256) {
    float s = 0.0f;
    for (int k = 0; k < KTOP; ++k) s += wk[k] * sess[topi[k] * 128 + d];
    outp[b * 128 + d] = s;
  }
}

// =======================================================================
extern "C" void kernel_launch(void* const* d_in, const int* in_sizes, int n_in,
                              void* d_out, int out_size, void* d_ws, size_t ws_size,
                              hipStream_t stream) {
  (void)in_sizes; (void)n_in; (void)out_size; (void)ws_size;
  const int* seq = (const int*)d_in[0];
  const float* emb = (const float*)d_in[2];
  const float* pos_emb = (const float*)d_in[3];
  enum { L_pos2item = 0, L_user2item, L_w_u_p, L_u_u_p, L_w_u_r, L_u_u_r, L_w_u, L_u_u,
         L_u_u_h, L_u_p_h, L_u_u_e, L_u_p_e, L_w_p_c, L_u_p_c, L_final2std_cur, L_v1_w,
         L_v1, L_v_h2std, L_final2std2_cur, L_final2std2_std, L_final2std3_cur, L_final2std3_std };
  auto Wl = [&](int li) { return (const float*)d_in[4 + 2 * li]; };
  auto Bl = [&](int li) { return (const float*)d_in[5 + 2 * li]; };
  const float* Wai = (const float*)d_in[48];
  const float* bai = (const float*)d_in[49];
  const float* Wao = (const float*)d_in[50];
  const float* bao = (const float*)d_in[51];
  const float* Wih = (const float*)d_in[52];
  const float* Whh = (const float*)d_in[53];
  const float* bih = (const float*)d_in[54];
  const float* bhh = (const float*)d_in[55];

  // ---------------- workspace carve ----------------
  char* base = (char*)d_ws;
  size_t off = 0;
  auto carve = [&](size_t bytes) -> void* {
    void* p = base + off;
    off = (off + bytes + 255) & ~(size_t)255;
    return p;
  };
  const size_t F32BL = (size_t)BLR * 128 * 4;
  const size_t BF16BL = (size_t)BLR * 128 * 2;
  float* GX = (float*)carve((size_t)BLR * 384 * 4);
  float* A0 = (float*)carve(F32BL);
  float* A1 = (float*)carve(F32BL);
  float* A2 = (float*)carve(F32BL);
  float* A3 = (float*)carve(F32BL);
  float* A4 = (float*)carve(F32BL);
  bf16_t* Bf0 = (bf16_t*)carve(BF16BL);
  bf16_t* Bf1 = (bf16_t*)carve(BF16BL);
  bf16_t* Bf2 = (bf16_t*)carve(BF16BL);
  bf16_t* Bf3 = (bf16_t*)carve(BF16BL);
  bf16_t* Bf4 = (bf16_t*)carve(BF16BL);
  bf16_t* Bf5 = (bf16_t*)carve(BF16BL);
  int* itempos = (int*)carve((size_t)BLR * 4);
  float* maskp = (float*)carve((size_t)BLR * 4);
  bf16_t* nowbf = (bf16_t*)carve((size_t)BB * 128 * 2);
  bf16_t* globf = (bf16_t*)carve((size_t)BB * 128 * 2);
  float* user_e = (float*)carve((size_t)BB * 128 * 4);
  bf16_t* userebf = (bf16_t*)carve((size_t)BB * 128 * 2);
  bf16_t* usernowbf = (bf16_t*)carve((size_t)BB * 128 * 2);
  float* user_p = (float*)carve((size_t)BB * 128 * 4);
  float* mu = (float*)carve(64 * 4);
  float* var = (float*)carve(64 * 4);
  float* svec = (float*)carve((size_t)BLR * 4);
  float* l1f = (float*)carve((size_t)BB * 128 * 4);
  bf16_t* l1bf = (bf16_t*)carve((size_t)BB * 128 * 2);
  bf16_t* vecbf = (bf16_t*)carve((size_t)BB * 128 * 2);
  float* VLf = (float*)carve((size_t)BB * 128 * 4);
  float* colsum = (float*)carve((size_t)LL * BB * 4);
  float* braw = (float*)carve((size_t)BLR * 4);
  float* beta = (float*)carve((size_t)BLR * 4);
  float* beta_v = (float*)carve(256);
  float* sessf[3];
  bf16_t* sessbf[3];
  for (int i = 0; i < 3; ++i) {
    sessf[i] = (float*)carve((size_t)BB * 128 * 4);
    sessbf[i] = (bf16_t*)carve((size_t)BB * 128 * 2);
  }
  bf16_t* sessn = (bf16_t*)carve((size_t)BB * 128 * 2);
  float* sim = (float*)carve((size_t)512 * 512 * 4);

  // alias plan (stream order makes reuse safe)
  float* embsf = A0;
  bf16_t* embsbf = Bf0;
  bf16_t* posbf = Bf1;
  bf16_t* posembbf = Bf2;
  float* user_r = A1;
  bf16_t* tmp1bf = Bf1;
  float* uwh = A2;
  bf16_t* uwbf = Bf3;
  float* fre_h = A0;
  float* fre_e = A1;
  float* pwh = A2;
  float* efpre = A3;
  float* F = A4;
  bf16_t* Fbf = Bf4;
  bf16_t* qpbf = Bf0;
  float* t1 = A0;
  float* ini_f = A1;
  bf16_t* keybf = Bf1;
  bf16_t* kpbf = Bf2;
  bf16_t* vpbf = Bf3;
  bf16_t* ctxbf = Bf5;
  float* attout = A2;

  auto gemm = [&](const bf16_t* X1, int rdiv1, const float* W1, const float* b1,
                  const bf16_t* X2, int rdiv2, const float* W2, const float* b2,
                  float* Yf, bf16_t* Ybf, int M, int N, int act) {
    int strips = (M / 16) * (N / 64);
    int blocks = (strips + 7) / 8;
    k_gemm<<<blocks, 256, 0, stream>>>(X1, rdiv1, W1, b1, X2, rdiv2, W2, b2, Yf, Ybf, M, N, act);
  };

  const int blocksBL = (int)(((long)BLR * 128 + 255) / 256);
  const int blocksB = (BB * 128 + 255) / 256;
  const size_t GRU_SMEM = 384 * 128 * 2 + 64 * 128 * 2 + 64 * 384 * 4;    // 212992
  const size_t ATTN_SMEM = (size_t)64 * 512 * 4 + (size_t)64 * 512 * 2;   // 196608

  // ---- prep + gather ----
  k_prep_idx<<<BB, 64, 0, stream>>>(seq, itempos, maskp);
  k_gather<<<blocksBL, 256, 0, stream>>>(seq, itempos, emb, pos_emb, embsf, embsbf, posbf);

  // ---- GRU ----
  gemm(embsbf, 1, Wih, bih, nullptr, 1, nullptr, nullptr, GX, nullptr, BLR, 384, 0);
  k_gru<<<8, 256, GRU_SMEM, stream>>>(GX, Whh, bhh, nowbf, globf);

  // ---- positional + user projections ----
  gemm(posbf, 1, Wl(L_pos2item), Bl(L_pos2item), nullptr, 1, nullptr, nullptr,
       nullptr, posembbf, BLR, 128, 0);
  gemm(globf, 1, Wl(L_user2item), Bl(L_user2item), nullptr, 1, nullptr, nullptr,
       user_e, userebf, BB, 128, 0);
  gemm(nowbf, 1, Wl(L_user2item), Bl(L_user2item), nullptr, 1, nullptr, nullptr,
       nullptr, usernowbf, BB, 128, 0);

  // ---- gated fusion ----
  gemm(userebf, 1, Wl(L_w_u_p), Bl(L_w_u_p), usernowbf, 1, Wl(L_u_u_p), Bl(L_u_u_p),
       user_p, nullptr, BB, 128, 1);
  gemm(userebf, LL, Wl(L_w_u_r), Bl(L_w_u_r), embsbf, 1, Wl(L_u_u_r), Bl(L_u_u_r),
       user_r, nullptr, BLR, 128, 1);
  k_ew_tmp1<<<blocksBL, 256, 0, stream>>>(user_p, embsf, tmp1bf);
  gemm(userebf, LL, Wl(L_w_u), Bl(L_w_u), tmp1bf, 1, Wl(L_u_u), Bl(L_u_u),
       uwh, nullptr, BLR, 128, 2);
  k_ew_uw<<<blocksBL, 256, 0, stream>>>(user_p, user_e, user_r, uwh, uwbf);
  gemm(uwbf, 1, Wl(L_u_u_h), Bl(L_u_u_h), posembbf, 1, Wl(L_u_p_h), Bl(L_u_p_h),
       fre_h, nullptr, BLR, 128, 1);
  gemm(uwbf, 1, Wl(L_u_u_e), Bl(L_u_u_e), posembbf, 1, Wl(L_u_p_e), Bl(L_u_p_e),
       fre_e, nullptr, BLR, 128, 1);
  gemm(uwbf, 1, Wl(L_w_p_c), Bl(L_w_p_c), posembbf, 1, Wl(L_u_p_c), Bl(L_u_p_c),
       pwh, nullptr, BLR, 128, 2);
  k_ew_final<<<blocksBL, 256, 0, stream>>>(fre_h, user_e, fre_e, pwh, efpre);

  // ---- batch norm ----
  k_bn_stats<<<LL, 256, 0, stream>>>(efpre, mu, var);
  k_bn_apply<<<blocksBL, 256, 0, stream>>>(efpre, mu, var, F, Fbf);

  // ---- shared std-layer inputs ----
  gemm(Fbf, 1, Wai, bai, nullptr, 1, nullptr, nullptr, nullptr, qpbf, BLR, 128, 0);
  gemm(Fbf, 1, Wl(L_v1_w), Bl(L_v1_w), nullptr, 1, nullptr, nullptr, t1, nullptr, BLR, 128, 2);
  k_rowdot<<<(BLR + 7) / 8, 256, 0, stream>>>(t1, Wl(L_v1), Bl(L_v1), svec, BLR);
  gemm(Fbf, 1, Wl(L_final2std_cur), Bl(L_final2std_cur), nullptr, 1, nullptr, nullptr,
       ini_f, keybf, BLR, 128, 0);
  k_l1<<<blocksB, 256, 0, stream>>>(maskp, svec, ini_f, l1f, l1bf);

  // ---- three std layers ----
  for (int layer = 0; layer < 3; ++layer) {
    const bf16_t* valsrc;
    if (layer == 0) {
      valsrc = l1bf;
    } else {
      int cur = (layer == 1) ? L_final2std2_cur : L_final2std3_cur;
      int stdw = (layer == 1) ? L_final2std2_std : L_final2std3_std;
      gemm(Fbf, 1, Wl(cur), Bl(cur), nullptr, 1, nullptr, nullptr,
           nullptr, keybf, BLR, 128, 0);
      gemm(sessbf[layer - 1], 1, Wl(stdw), Bl(stdw), nullptr, 1, nullptr, nullptr,
           nullptr, vecbf, BB, 128, 0);
      valsrc = vecbf;
    }
    gemm(valsrc, 1, Wai + 2 * 128 * 128, nullptr, nullptr, 1, nullptr, nullptr,
         VLf, nullptr, BB, 128, 0);
    k_vp<<<blocksBL, 256, 0, stream>>>(maskp, VLf, bai + 256, vpbf);
    gemm(keybf, 1, Wai + 128 * 128, bai + 128, nullptr, 1, nullptr, nullptr,
         nullptr, kpbf, BLR, 128, 0);
    hipMemsetAsync(colsum, 0, (size_t)LL * BB * 4, stream);
    k_attn<<<LL * NHEADS * 8, 256, ATTN_SMEM, stream>>>(qpbf, kpbf, vpbf, ctxbf, colsum);
    gemm(ctxbf, 1, Wao, bao, nullptr, 1, nullptr, nullptr, attout, nullptr, BLR, 128, 0);
    k_beta_pre<<<(BLR + 7) / 8, 256, 0, stream>>>(attout, F, colsum,
                                                  Wl(L_v_h2std), Bl(L_v_h2std), braw);
    k_beta_softmax<<<BB, 64, 0, stream>>>(braw, beta, beta_v);
    k_sess<<<blocksB, 256, 0, stream>>>(beta, beta_v, F, sessf[layer], sessbf[layer]);
  }

  // ---- neighbor mixing ----
  k_norm<<<(BB + 7) / 8, 256, 0, stream>>>(sessf[2], sessn);
  k_sim<<<(1024 + 7) / 8, 256, 0, stream>>>(sessn, sim);
  k_topk<<<BB, 256, 0, stream>>>(sim, sessf[2], (float*)d_out);
}
